// SelfAttention_50276887167580
// MI455X (gfx1250) — matrix operand order repaired and checked by reading, not compiled, not run on hardware
//
#include <hip/hip_runtime.h>

#define DM   1024   // d_model
#define NH   16     // heads
#define DH   64     // head dim
#define NB   2      // batch
#define SL   2048   // sequence length

typedef __bf16 bf16_t;
typedef bf16_t v16bf __attribute__((ext_vector_type(16)));
typedef bf16_t v8bf  __attribute__((ext_vector_type(8)));
typedef float  v8f   __attribute__((ext_vector_type(8)));

// ---------------------------------------------------------------------------
// WMMA helpers (CDNA5 gfx1250, wave32)
// ---------------------------------------------------------------------------
static __device__ __forceinline__ v8f v8f_zero() {
  v8f z;
#pragma unroll
  for (int i = 0; i < 8; ++i) z[i] = 0.0f;
  return z;
}

static __device__ __forceinline__ v8f wmma_bf16(v16bf a, v16bf b, v8f c) {
  // D = A(16x32) * B(32x16) + C(16x16 f32)
  return __builtin_amdgcn_wmma_f32_16x16x32_bf16(false, a, false, b,
                                                 (short)0, c, false, false);
}

// A-matrix fragment (16x32 bf16, MxK). Per ISA layout:
// lanes 0-15: row M=lane, K = {0..7, 16..23}; lanes 16-31: row M=lane-16,
// K = {8..15, 24..31}.  Both halves are contiguous 8-element (16B) chunks.
static __device__ __forceinline__ v16bf frag_a(const bf16_t* __restrict__ base,
                                               int ld, int k0, int lane) {
  const int row = lane & 15;
  const int kb  = (lane & 16) ? 8 : 0;
  const bf16_t* p = base + (size_t)row * ld + (k0 + kb);
  v8bf lo = *reinterpret_cast<const v8bf*>(p);        // K = k0+kb .. +7
  v8bf hi = *reinterpret_cast<const v8bf*>(p + 16);   // K = k0+16+kb .. +7
  return __builtin_shufflevector(lo, hi, 0, 1, 2, 3, 4, 5, 6, 7,
                                 8, 9, 10, 11, 12, 13, 14, 15);
}

// B-matrix fragment (32x16 bf16, KxN) read from a row-major "B-source" whose
// ROWS are indexed by N (K contiguous):  lanes 0-15: col N=lane, K={0..7, 16..23};
// lanes 16-31: col N=lane-16, K={8..15, 24..31}.  Two 8-element (16B) chunks.
static __device__ __forceinline__ v16bf frag_b(const bf16_t* __restrict__ base,
                                               int ld, int k0, int lane) {
  const int n  = lane & 15;
  const int kb = (lane & 16) ? 8 : 0;
  const bf16_t* p = base + (size_t)n * ld + (k0 + kb);
  v8bf lo = *reinterpret_cast<const v8bf*>(p);
  v8bf hi = *reinterpret_cast<const v8bf*>(p + 16);
  return __builtin_shufflevector(lo, hi, 0, 1, 2, 3, 4, 5, 6, 7,
                                 8, 9, 10, 11, 12, 13, 14, 15);
}

static __device__ __forceinline__ float hmax16(float v) {
  v = fmaxf(v, __shfl_xor(v, 1, 16));
  v = fmaxf(v, __shfl_xor(v, 2, 16));
  v = fmaxf(v, __shfl_xor(v, 4, 16));
  v = fmaxf(v, __shfl_xor(v, 8, 16));
  return v;
}
static __device__ __forceinline__ float hsum16(float v) {
  v += __shfl_xor(v, 1, 16);
  v += __shfl_xor(v, 2, 16);
  v += __shfl_xor(v, 4, 16);
  v += __shfl_xor(v, 8, 16);
  return v;
}

// ---------------------------------------------------------------------------
// f32 -> bf16 elementwise convert
// ---------------------------------------------------------------------------
__global__ __launch_bounds__(256) void cvt_bf16_kernel(
    const float* __restrict__ in, bf16_t* __restrict__ out, int n) {
  int i = blockIdx.x * 256 + threadIdx.x;
  if (i < n) out[i] = (bf16_t)in[i];
}

// ---------------------------------------------------------------------------
// x [b][i][l] f32  ->  xT [b][l][i] bf16   (LDS-tiled transpose)
// ---------------------------------------------------------------------------
__global__ __launch_bounds__(256) void transpose_cvt_kernel(
    const float* __restrict__ x, bf16_t* __restrict__ xT) {
  __shared__ float tile[32][33];
  const int l0 = blockIdx.x * 32;
  const int i0 = blockIdx.y * 32;
  const int b  = blockIdx.z;
  const int tx = threadIdx.x, ty = threadIdx.y;
#pragma unroll
  for (int j = 0; j < 4; ++j) {
    int i = i0 + ty + 8 * j;
    tile[ty + 8 * j][tx] = x[(size_t)b * DM * SL + (size_t)i * SL + l0 + tx];
  }
  __syncthreads();
#pragma unroll
  for (int j = 0; j < 4; ++j) {
    int l = l0 + ty + 8 * j;
    xT[(size_t)b * SL * DM + (size_t)l * DM + i0 + tx] = (bf16_t)tile[tx][ty + 8 * j];
  }
}

// ---------------------------------------------------------------------------
// Universal NT GEMM:  C[m][n] = sum_k A[m][k] * B[n][k]
// Block = 256 threads = 8 waves, 128x256 block tile; wave = 64x64
// (4x4 WMMA tiles -> 16 WMMAs per 8 operand fragments per k-step).
// All operands loaded straight from global memory (L2-resident working set).
// ---------------------------------------------------------------------------
__global__ __launch_bounds__(256) void gemm_nt_kernel(
    const bf16_t* __restrict__ A, size_t a_bs,
    const bf16_t* __restrict__ B, size_t b_bs,
    void* __restrict__ C, size_t c_bs,
    int K, int lda, int ldb, int ldc, int c_is_f32) {
  const int lane = threadIdx.x & 31;
  const int wave = threadIdx.x >> 5;
  const int wm = wave & 1;        // 0..1  -> 64-row group
  const int wn = wave >> 1;       // 0..3  -> 64-col group
  const int bz = blockIdx.z;

  const bf16_t* Ab = A + (size_t)bz * a_bs;
  const bf16_t* Bb = B + (size_t)bz * b_bs;
  const int m0 = blockIdx.x * 128 + wm * 64;
  const int n0 = blockIdx.y * 256 + wn * 64;

  const bf16_t* arow[4];
  const bf16_t* brow[4];
#pragma unroll
  for (int i = 0; i < 4; ++i) arow[i] = Ab + (size_t)(m0 + 16 * i) * lda;
#pragma unroll
  for (int j = 0; j < 4; ++j) brow[j] = Bb + (size_t)(n0 + 16 * j) * ldb;

  v8f acc[4][4];
#pragma unroll
  for (int i = 0; i < 4; ++i)
#pragma unroll
    for (int j = 0; j < 4; ++j) acc[i][j] = v8f_zero();

  for (int k0 = 0; k0 < K; k0 += 32) {
    v16bf a[4], b[4];
#pragma unroll
    for (int i = 0; i < 4; ++i) a[i] = frag_a(arow[i], lda, k0, lane);
#pragma unroll
    for (int j = 0; j < 4; ++j) b[j] = frag_b(brow[j], ldb, k0, lane);
#pragma unroll
    for (int i = 0; i < 4; ++i)
#pragma unroll
      for (int j = 0; j < 4; ++j) acc[i][j] = wmma_bf16(a[i], b[j], acc[i][j]);
  }

  const int rlo = (lane & 16) ? 8 : 0;
  const int cl  = lane & 15;
  if (c_is_f32) {
    float* Cf = (float*)C + (size_t)bz * c_bs;
#pragma unroll
    for (int i = 0; i < 4; ++i)
#pragma unroll
      for (int j = 0; j < 4; ++j)
#pragma unroll
        for (int r = 0; r < 8; ++r)
          Cf[(size_t)(m0 + 16 * i + r + rlo) * ldc + (n0 + 16 * j + cl)] = acc[i][j][r];
  } else {
    bf16_t* Cb = (bf16_t*)C + (size_t)bz * c_bs;
#pragma unroll
    for (int i = 0; i < 4; ++i)
#pragma unroll
      for (int j = 0; j < 4; ++j)
#pragma unroll
        for (int r = 0; r < 8; ++r)
          Cb[(size_t)(m0 + 16 * i + r + rlo) * ldc + (n0 + 16 * j + cl)] =
              (bf16_t)acc[i][j][r];
  }
}

// ---------------------------------------------------------------------------
// Flash attention. Grid = (SL/128, NH, NB), block = 128 threads (4 waves).
// Each wave owns 32 output rows (two 16-row groups); inner loop walks 64 key
// columns at a time.  Q-side and V-side fragments are loaded once per step
// and reused by both row groups (2x arithmetic intensity).
//   Kt (== reference Kp) : [b][l][o]  bf16  (A-source, d contiguous)
//   Qt (== reference Qp) : [b][m][o]  bf16  (B-source for S, d contiguous)
//   Vt (== reference Vp) : [b][o][l]  bf16  (B-source for O, m contiguous)
//   Ot                   : [b][l][o]  bf16
// ---------------------------------------------------------------------------
__global__ __launch_bounds__(128) void attn_kernel(
    const bf16_t* __restrict__ Kt, const bf16_t* __restrict__ Qt,
    const bf16_t* __restrict__ Vt, const int* __restrict__ mask,
    bf16_t* __restrict__ Ot) {
  __shared__ bf16_t Pl[4][32][72];   // per-wave P tile, pad keeps 16B chunks
  const int lane = threadIdx.x & 31;
  const int wave = threadIdx.x >> 5;
  const int b = blockIdx.z, h = blockIdx.y;
  const int rowbase = blockIdx.x * 128 + wave * 32;   // 32 rows per wave

  const bf16_t* Qb   = Qt + (size_t)b * SL * DM + h * DH;
  const bf16_t* Vb   = Vt + ((size_t)b * DM + h * DH) * SL;
  const int*    mrow = mask + (size_t)b * SL;

  // Query-side A fragments are loop-invariant (K = DH = 64 -> two per group).
  v16bf aq[2][2];
#pragma unroll
  for (int g = 0; g < 2; ++g) {
    const bf16_t* Abase = Kt + ((size_t)b * SL + rowbase + 16 * g) * DM + h * DH;
    aq[g][0] = frag_a(Abase, DM, 0, lane);
    aq[g][1] = frag_a(Abase, DM, 32, lane);
  }

  v8f oacc[2][4];
  float rowm[2][8], rowl[2][8];
#pragma unroll
  for (int g = 0; g < 2; ++g) {
#pragma unroll
    for (int f = 0; f < 4; ++f) oacc[g][f] = v8f_zero();
#pragma unroll
    for (int r = 0; r < 8; ++r) { rowm[g][r] = -3.0e38f; rowl[g][r] = 0.0f; }
  }

  const int rlo = (lane & 16) ? 8 : 0;
  const int cl  = lane & 15;

  for (int m0 = 0; m0 < SL; m0 += 64) {
    // ----- S = Kp_tile . Qp_tile^T  (32 x 64) -----
    v8f s[2][4];
#pragma unroll
    for (int g = 0; g < 2; ++g)
#pragma unroll
      for (int j = 0; j < 4; ++j) s[g][j] = v8f_zero();
#pragma unroll
    for (int j = 0; j < 4; ++j) {
      const bf16_t* qr = Qb + (size_t)(m0 + 16 * j) * DM;
      const v16bf bq0 = frag_b(qr, DM, 0, lane);
      const v16bf bq1 = frag_b(qr, DM, 32, lane);
#pragma unroll
      for (int g = 0; g < 2; ++g) {
        s[g][j] = wmma_bf16(aq[g][0], bq0, s[g][j]);
        s[g][j] = wmma_bf16(aq[g][1], bq1, s[g][j]);
      }
    }
    // ----- scale 1/sqrt(64) + key mask -----
#pragma unroll
    for (int j = 0; j < 4; ++j) {
      const int mv = mrow[m0 + 16 * j + cl];
#pragma unroll
      for (int g = 0; g < 2; ++g)
#pragma unroll
        for (int r = 0; r < 8; ++r)
          s[g][j][r] = mv ? s[g][j][r] * 0.125f : -1.0e30f;
    }
    // ----- online softmax per row group -----
#pragma unroll
    for (int g = 0; g < 2; ++g) {
      float mnew[8], alpha[8];
#pragma unroll
      for (int r = 0; r < 8; ++r) {
        float v = fmaxf(fmaxf(s[g][0][r], s[g][1][r]),
                        fmaxf(s[g][2][r], s[g][3][r]));
        mnew[r] = fmaxf(rowm[g][r], hmax16(v));
        alpha[r] = __expf(rowm[g][r] - mnew[r]);
        rowm[g][r] = mnew[r];
      }
      float rs[8];
#pragma unroll
      for (int r = 0; r < 8; ++r) rs[r] = 0.0f;
#pragma unroll
      for (int j = 0; j < 4; ++j)
#pragma unroll
        for (int r = 0; r < 8; ++r) {
          float p = __expf(s[g][j][r] - mnew[r]);
          s[g][j][r] = p;
          rs[r] += p;
        }
#pragma unroll
      for (int r = 0; r < 8; ++r)
        rowl[g][r] = rowl[g][r] * alpha[r] + hsum16(rs[r]);
#pragma unroll
      for (int f = 0; f < 4; ++f)
#pragma unroll
        for (int r = 0; r < 8; ++r) oacc[g][f][r] *= alpha[r];

      // P (C-layout f32) -> LDS bf16 (A-layout source)
#pragma unroll
      for (int j = 0; j < 4; ++j)
#pragma unroll
        for (int r = 0; r < 8; ++r)
          Pl[wave][16 * g + r + rlo][16 * j + cl] = (bf16_t)s[g][j][r];
    }
    // wave-private region; DS is in-order per wave, but be explicit:
    asm volatile("s_wait_dscnt 0" ::: "memory");

    v16bf p[2][2];
#pragma unroll
    for (int g = 0; g < 2; ++g) {
      p[g][0] = frag_a(&Pl[wave][16 * g][0], 72, 0, lane);
      p[g][1] = frag_a(&Pl[wave][16 * g][0], 72, 32, lane);
    }

    // ----- O += P . V_tile  (32 x 64) -----
#pragma unroll
    for (int f = 0; f < 4; ++f) {
      const bf16_t* vr = Vb + (size_t)(16 * f) * SL;
      const v16bf vb0 = frag_b(vr, SL, m0, lane);
      const v16bf vb1 = frag_b(vr, SL, m0 + 32, lane);
#pragma unroll
      for (int g = 0; g < 2; ++g) {
        oacc[g][f] = wmma_bf16(p[g][0], vb0, oacc[g][f]);
        oacc[g][f] = wmma_bf16(p[g][1], vb1, oacc[g][f]);
      }
    }
  }

  // ----- normalize and store Ot[b][l][h*64+d] -----
#pragma unroll
  for (int g = 0; g < 2; ++g)
#pragma unroll
    for (int f = 0; f < 4; ++f)
#pragma unroll
      for (int r = 0; r < 8; ++r) {
        const float val = oacc[g][f][r] / rowl[g][r];
        const int row = rowbase + 16 * g + r + rlo;
        Ot[((size_t)b * SL + row) * DM + h * DH + 16 * f + cl] = (bf16_t)val;
      }
}

// ---------------------------------------------------------------------------
// Host orchestration
// ---------------------------------------------------------------------------
extern "C" void kernel_launch(void* const* d_in, const int* in_sizes, int n_in,
                              void* d_out, int out_size, void* d_ws, size_t ws_size,
                              hipStream_t stream) {
  const float* x    = (const float*)d_in[0];
  const int*   mask = (const int*)d_in[1];
  const float* W[4] = {(const float*)d_in[2], (const float*)d_in[3],
                       (const float*)d_in[4], (const float*)d_in[5]};

  char* ws = (char*)d_ws;
  size_t off = 0;
  auto carve = [&](size_t bytes) -> void* {
    void* p = ws + off;
    off += (bytes + 255) & ~(size_t)255;
    return p;
  };
  bf16_t* Wb[4];
  for (int k = 0; k < 4; ++k) Wb[k] = (bf16_t*)carve((size_t)DM * DM * 2);
  bf16_t* xT = (bf16_t*)carve((size_t)NB * SL * DM * 2);
  bf16_t* Kt = (bf16_t*)carve((size_t)NB * SL * DM * 2);
  bf16_t* Qt = (bf16_t*)carve((size_t)NB * SL * DM * 2);
  bf16_t* Vt = (bf16_t*)carve((size_t)NB * DM * SL * 2);
  bf16_t* Ot = (bf16_t*)carve((size_t)NB * SL * DM * 2);

  // 1) weights f32 -> bf16
  const int nW = DM * DM;
  for (int k = 0; k < 4; ++k)
    cvt_bf16_kernel<<<dim3((nW + 255) / 256), dim3(256), 0, stream>>>(W[k], Wb[k], nW);

  // 2) x -> xT (bf16, [b][l][i])
  transpose_cvt_kernel<<<dim3(SL / 32, DM / 32, NB), dim3(32, 8), 0, stream>>>(x, xT);

  // 3) Kp, Qp: C[l][o] = xT[l][:] . W[o][:]   -> [b][l][o] bf16
  dim3 gLD(SL / 128, DM / 256, NB);
  gemm_nt_kernel<<<gLD, 256, 0, stream>>>(xT, (size_t)SL * DM, Wb[0], 0,
                                          Kt, (size_t)SL * DM, DM, DM, DM, DM, 0);
  gemm_nt_kernel<<<gLD, 256, 0, stream>>>(xT, (size_t)SL * DM, Wb[1], 0,
                                          Qt, (size_t)SL * DM, DM, DM, DM, DM, 0);

  // 4) Vp: C[o][l] = W2[o][:] . xT[l][:]      -> [b][o][l] bf16
  dim3 gDL(DM / 128, SL / 256, NB);
  gemm_nt_kernel<<<gDL, 256, 0, stream>>>(Wb[2], 0, xT, (size_t)SL * DM,
                                          Vt, (size_t)DM * SL, DM, DM, DM, SL, 0);

  // 5) flash attention -> Ot [b][l][o] bf16
  attn_kernel<<<dim3(SL / 128, NH, NB), dim3(128), 0, stream>>>(Kt, Qt, Vt, mask, Ot);

  // 6) final projection: out[b][o][l] = W3[o][:] . Ot[l][:]  (f32 to d_out)
  gemm_nt_kernel<<<gDL, 256, 0, stream>>>(Wb[3], 0, Ot, (size_t)SL * DM,
                                          d_out, (size_t)DM * SL, DM, DM, DM, SL, 1);
}
